// MetaBaseline_40321152975640
// MI455X (gfx1250) — matrix operand order repaired and checked by reading, not compiled, not run on hardware
//
#include <hip/hip_runtime.h>
#include <hip/hip_bf16.h>
#include <math.h>

typedef __attribute__((ext_vector_type(16))) _Float16 v16h;
typedef __attribute__((ext_vector_type(8)))  float    v8f;

#define B_    8
#define W_    5
#define S_    5
#define C_    640
#define HW_   100
#define Q_    75
#define ND_   500          // S*hw support descriptors
#define NDP_  512          // padded to 32 n-tiles of 16
#define KS_   32           // WMMA K step (f16)
#define KITER_ (C_ / KS_)  // 20 k-steps
#define MTA_  4            // m-tiles register-blocked per pass (64 rows)
#define NPASS_ 2           // 2 passes x 64 rows = 128 rows (>=100, masked)

__device__ __forceinline__ void wait_async0() {
#if __has_builtin(__builtin_amdgcn_s_wait_asynccnt)
  __builtin_amdgcn_s_wait_asynccnt(0);
#else
  asm volatile("s_wait_asynccnt 0x0" ::: "memory");
#endif
}

__device__ __forceinline__ void async_b128_to_lds(unsigned lds_addr, const void* gaddr) {
  // GLOBAL_LOAD_ASYNC_TO_LDS_B128: vdst = LDS byte address VGPR, vaddr = 64-bit global addr
  asm volatile("global_load_async_to_lds_b128 %0, %1, off"
               :: "v"(lds_addr), "v"(gaddr) : "memory");
}

// ---------------- Kernel 1a: query descriptors, L2-normalize over C, emit f16 [bq, p, c]
__global__ void qd_norm_kernel(const float* __restrict__ xq, _Float16* __restrict__ qd) {
  int row = blockIdx.x;                // bq*HW_ + p
  int p   = row % HW_;
  int bq  = row / HW_;
  const float* src = xq + (size_t)bq * C_ * HW_ + p;   // element c at src[c*HW_]
  __shared__ float red[128];
  float ss = 0.f;
  for (int c = threadIdx.x; c < C_; c += 128) {
    float v = src[(size_t)c * HW_];
    ss += v * v;
  }
  red[threadIdx.x] = ss;
  __syncthreads();
  for (int s = 64; s > 0; s >>= 1) {
    if (threadIdx.x < s) red[threadIdx.x] += red[threadIdx.x + s];
    __syncthreads();
  }
  float rn = rsqrtf(fmaxf(red[0], 1e-30f));
  _Float16* dst = qd + (size_t)row * C_;
  for (int c = threadIdx.x; c < C_; c += 128)
    dst[c] = (_Float16)(src[(size_t)c * HW_] * rn);
}

// ---------------- Kernel 1b: query GAP mean over hw -> qmean[bq, c]
__global__ void qmean_kernel(const float* __restrict__ xq, float* __restrict__ qmean) {
  int idx = blockIdx.x * 256 + threadIdx.x;   // bq*C_ + c
  if (idx >= B_ * Q_ * C_) return;
  const float* src = xq + (size_t)idx * HW_;
  float s = 0.f;
  for (int p = 0; p < HW_; ++p) s += src[p];
  qmean[idx] = s * (1.f / HW_);
}

// ---------------- Kernel 2: support: normalize per (b,w,c) across d; emit f16 [bw, d(512), c]; proto = mean
__global__ void sup_norm_kernel(const float* __restrict__ xs, _Float16* __restrict__ sup,
                                float* __restrict__ proto) {
  int idx = blockIdx.x;        // bw*C_ + c
  int c   = idx % C_;
  int bw  = idx / C_;
  const float* base = xs + ((size_t)bw * S_ * C_ + c) * HW_;  // element (s,p) at base[s*C_*HW_ + p]
  __shared__ float rs[128], r2[128];
  float s = 0.f, sq = 0.f;
  for (int d = threadIdx.x; d < ND_; d += 128) {
    int si = d / HW_, p = d % HW_;
    float v = base[(size_t)si * C_ * HW_ + p];
    s += v; sq += v * v;
  }
  rs[threadIdx.x] = s; r2[threadIdx.x] = sq;
  __syncthreads();
  for (int st = 64; st > 0; st >>= 1) {
    if (threadIdx.x < st) { rs[threadIdx.x] += rs[threadIdx.x + st]; r2[threadIdx.x] += r2[threadIdx.x + st]; }
    __syncthreads();
  }
  float rn = rsqrtf(fmaxf(r2[0], 1e-30f));
  _Float16* dst = sup + (size_t)bw * NDP_ * C_ + c;           // element d at dst[d*C_]
  for (int d = threadIdx.x; d < ND_; d += 128) {
    int si = d / HW_, p = d % HW_;
    float v = base[(size_t)si * C_ * HW_ + p];
    dst[(size_t)d * C_] = (_Float16)(v * rn);
  }
  for (int d = ND_ + threadIdx.x; d < NDP_; d += 128)
    dst[(size_t)d * C_] = (_Float16)0.f;                      // zero pad rows 500..511
  if (threadIdx.x == 0) proto[idx] = rs[0] * (1.f / ND_);
}

// ---------------- Main: per (b,q,w): async double-buffered WMMA GEMM fused with top-k + cosine
__global__ void __launch_bounds__(256) dn4_main_kernel(
    const _Float16* __restrict__ qd, const _Float16* __restrict__ sup,
    const float* __restrict__ qmean, const float* __restrict__ proto,
    const float* __restrict__ r_cos, const float* __restrict__ r_dn4,
    const int* __restrict__ nk_ptr, float* __restrict__ out)
{
  // 2 x 32KB B buffers (buffer 0 reused as 16x512 f32 Sim tile in epilogue)
  __shared__ __align__(16) unsigned char smemB[2][NDP_ * KS_ * 2];
  // 8KB: double-buffered A tiles [2][64][32] f16  UNION  cand[16][128] f32 (epilogue only)
  __shared__ __align__(16) unsigned char smemA[2 * MTA_ * 16 * KS_ * 2];
  __shared__ float dn4_acc;

  float (*Sim)[NDP_]  = reinterpret_cast<float (*)[NDP_]>(smemB[0]);
  float (*cand)[128]  = reinterpret_cast<float (*)[128]>(smemA);

  const int bid = blockIdx.x;
  const int w   = bid % W_;
  const int q   = (bid / W_) % Q_;
  const int b   = bid / (W_ * Q_);
  const int bq  = b * Q_ + q;
  const int bw  = b * W_ + w;

  const int tid  = threadIdx.x;
  const int lane = tid & 31;
  const int wv   = tid >> 5;     // wave id 0..7 -> owns columns [wv*64, wv*64+64)
  const int hf   = lane >> 4;    // lane half (wave32)
  const int l16  = lane & 15;

  int nk = *nk_ptr; if (nk < 1) nk = 1; if (nk > 8) nk = 8;
  if (tid == 0) dn4_acc = 0.f;

  const _Float16* qbase = qd  + (size_t)bq * HW_  * C_;
  const _Float16* sbase = sup + (size_t)bw * NDP_ * C_;

  for (int pass = 0; pass < NPASS_; ++pass) {
    const int mbase = pass * (MTA_ * 16);      // row base 0 / 64

    // stage(k0 -> buffer sel): async B chunk (512x32 f16) + A tiles (64x32 f16, zero-masked)
    auto stage = [&](int k0, int sel) {
      _Float16 (*Bsb)[KS_] = reinterpret_cast<_Float16 (*)[KS_]>(smemB[sel]);
      const _Float16* g0 = sbase + (size_t)tid * C_ + k0;
      const _Float16* g1 = sbase + (size_t)(tid + 256) * C_ + k0;
      unsigned l0 = (unsigned)(size_t)&Bsb[tid][0];
      unsigned l1 = (unsigned)(size_t)&Bsb[tid + 256][0];
      #pragma unroll
      for (int sgm = 0; sgm < 4; ++sgm) {
        async_b128_to_lds(l0 + sgm * 16, g0 + sgm * 8);
        async_b128_to_lds(l1 + sgm * 16, g1 + sgm * 8);
      }
      _Float16 (*Asb)[KS_] = reinterpret_cast<_Float16 (*)[KS_]>(smemA) + sel * (MTA_ * 16);
      int row = tid >> 2, seg = tid & 3;
      int p = mbase + row;
      uint4 v = {0u, 0u, 0u, 0u};
      if (p < HW_)
        v = *reinterpret_cast<const uint4*>(qbase + (size_t)p * C_ + k0 + seg * 8);
      *reinterpret_cast<uint4*>(&Asb[row][seg * 8]) = v;
    };

    v8f acc[MTA_][4];
    #pragma unroll
    for (int mi = 0; mi < MTA_; ++mi)
      #pragma unroll
      for (int t = 0; t < 4; ++t) acc[mi][t] = v8f{};

    // pipeline prologue: fill buffer 0
    stage(0, 0);
    wait_async0();
    __syncthreads();

    for (int ki = 0; ki < KITER_; ++ki) {
      const int cur = ki & 1;
      // prefetch next chunk into the other buffer while we compute (async DMA overlaps WMMA)
      if (ki + 1 < KITER_) stage((ki + 1) * KS_, cur ^ 1);

      _Float16 (*Bsb)[KS_] = reinterpret_cast<_Float16 (*)[KS_]>(smemB[cur]);
      _Float16 (*Asb)[KS_] = reinterpret_cast<_Float16 (*)[KS_]>(smemA) + cur * (MTA_ * 16);

      // build 4 A operands (ISA 16-bit A 16x32 layout)
      v16h a[MTA_];
      const int kb = hf * 8;
      #pragma unroll
      for (int mi = 0; mi < MTA_; ++mi) {
        const int arow = mi * 16 + l16;
        #pragma unroll
        for (int j = 0; j < 8; ++j) {
          int kk = (j < 4) ? (kb + 2 * j) : (kb + 16 + 2 * (j - 4));
          a[mi][2 * j]     = Asb[arow][kk];
          a[mi][2 * j + 1] = Asb[arow][kk + 1];
        }
      }
      // 4 n-tiles x 4 m-tiles = 16 WMMAs per wave per k-step
      #pragma unroll
      for (int t = 0; t < 4; ++t) {
        int n = wv * 64 + t * 16 + l16;
        v16h bm;
        #pragma unroll
        for (int v = 0; v < 8; ++v) {
          bm[2 * v]     = Bsb[n][(v < 4) ? (hf * 8 + 2 * v) : (hf * 8 + 16 + 2 * (v - 4))];
          bm[2 * v + 1] = Bsb[n][((v < 4) ? (hf * 8 + 2 * v) : (hf * 8 + 16 + 2 * (v - 4))) + 1];
        }
        #pragma unroll
        for (int mi = 0; mi < MTA_; ++mi)
          acc[mi][t] = __builtin_amdgcn_wmma_f32_16x16x32_f16(
              false, a[mi], false, bm, (short)0, acc[mi][t], false, false);
      }

      wait_async0();        // my prefetch into buffer cur^1 landed
      __syncthreads();      // everyone done reading cur & staging cur^1
    }

    // ---- epilogue: one 16x512 m-tile at a time (Sim reuses B buffer 0, cand reuses A buffers)
    #pragma unroll
    for (int mi = 0; mi < MTA_; ++mi) {
      const int prow0 = mbase + mi * 16;
      __syncthreads();
      #pragma unroll
      for (int t = 0; t < 4; ++t)
        #pragma unroll
        for (int r = 0; r < 8; ++r)
          Sim[r + 8 * hf][wv * 64 + t * 16 + l16] = acc[mi][t][r];
      __syncthreads();

      // stage-1 top-k: 16 threads/row, each scans 32 columns keeping sorted top-8
      {
        int row = tid >> 4;        // 0..15
        int seg = tid & 15;        // 0..15
        int p = prow0 + row;
        float best[8];
        #pragma unroll
        for (int i = 0; i < 8; ++i) best[i] = -3.0e38f;
        if (p < HW_) {
          for (int j = 0; j < 32; ++j) {
            int col = seg * 32 + j;
            float v = (col < ND_) ? Sim[row][col] : -3.0e38f;
            #pragma unroll
            for (int i = 0; i < 8; ++i) {
              if (v > best[i]) { float tmp = best[i]; best[i] = v; v = tmp; }
            }
          }
        }
        #pragma unroll
        for (int i = 0; i < 8; ++i) cand[row][seg * 8 + i] = best[i];
      }
      __syncthreads();
      // stage-2 merge: one thread per row selects top-nk of 128 candidates
      if (tid < 16) {
        int p = prow0 + tid;
        if (p < HW_) {
          float s = 0.f;
          for (int j = 0; j < nk; ++j) {
            float mx = -3.0e38f; int mxi = 0;
            for (int i = 0; i < 128; ++i) {
              float v = cand[tid][i];
              if (v > mx) { mx = v; mxi = i; }
            }
            s += mx;
            cand[tid][mxi] = -3.0e38f;
          }
          atomicAdd(&dn4_acc, s);       // ds_add_f32
        }
      }
    }
    __syncthreads();   // protect Sim/cand buffers before next pass restages them
  }

  // ---- cosine-prototype branch + final blend (wave 0)
  if (tid < 32) {
    const float* qm = qmean + (size_t)bq * C_;
    const float* pr = proto + (size_t)bw * C_;
    float dqp = 0.f, dqq = 0.f, dpp = 0.f;
    for (int c = lane; c < C_; c += 32) {
      float av = qm[c], pv = pr[c];
      dqp += av * pv; dqq += av * av; dpp += pv * pv;
    }
    #pragma unroll
    for (int off = 16; off > 0; off >>= 1) {
      dqp += __shfl_down(dqp, off, 32);
      dqq += __shfl_down(dqq, off, 32);
      dpp += __shfl_down(dpp, off, 32);
    }
    if (lane == 0) {
      float cosv = dqp * rsqrtf(fmaxf(dqq * dpp, 1e-30f));
      out[bid] = r_cos[0] * cosv + r_dn4[0] * (dn4_acc / (float)nk);
    }
  }
}

extern "C" void kernel_launch(void* const* d_in, const int* in_sizes, int n_in,
                              void* d_out, int out_size, void* d_ws, size_t ws_size,
                              hipStream_t stream) {
  const float* x_shot  = (const float*)d_in[0];
  const float* x_query = (const float*)d_in[1];
  const float* r_cos   = (const float*)d_in[2];
  const float* r_dn4   = (const float*)d_in[3];
  const int*   nk      = (const int*)d_in[4];
  float* out = (float*)d_out;

  char* ws = (char*)d_ws;
  size_t off = 0;
  auto alloc = [&](size_t bytes) -> void* {
    void* p = ws + off;
    off = (off + bytes + 255) & ~(size_t)255;
    return p;
  };
  _Float16* qd    = (_Float16*)alloc((size_t)B_ * Q_ * HW_  * C_ * sizeof(_Float16)); // 76.8 MB
  _Float16* sup   = (_Float16*)alloc((size_t)B_ * W_ * NDP_ * C_ * sizeof(_Float16)); // 26.2 MB
  float*    qmean = (float*)   alloc((size_t)B_ * Q_ * C_ * sizeof(float));
  float*    proto = (float*)   alloc((size_t)B_ * W_ * C_ * sizeof(float));

  qd_norm_kernel <<<B_ * Q_ * HW_, 128, 0, stream>>>(x_query, qd);
  qmean_kernel   <<<(B_ * Q_ * C_ + 255) / 256, 256, 0, stream>>>(x_query, qmean);
  sup_norm_kernel<<<B_ * W_ * C_, 128, 0, stream>>>(x_shot, sup, proto);
  dn4_main_kernel<<<B_ * Q_ * W_, 256, 0, stream>>>(qd, sup, qmean, proto, r_cos, r_dn4, nk, out);
}
